// InteractionPruningEluLayerV2_72378788872997
// MI455X (gfx1250) — matrix operand order repaired and checked by reading, not compiled, not run on hardware
//
#include <hip/hip_runtime.h>
#include <math.h>

typedef __attribute__((ext_vector_type(16))) __bf16 v16bf;
typedef __attribute__((ext_vector_type(8)))  float  v8f;

#define NN 256   // rows per batch
#define DD 128   // feature dim

// ---------------------------------------------------------------------------
// WMMA wrapper: D = A(16x32 bf16) * B(32x16 bf16) + C(16x16 f32)
// ---------------------------------------------------------------------------
__device__ __forceinline__ v8f wmma_bf16(v16bf a, v16bf b, v8f c) {
  return __builtin_amdgcn_wmma_f32_16x16x32_bf16(
      /*neg_a=*/false, a, /*neg_b=*/false, b,
      /*c_mod=*/(short)0, c, /*reuse_a=*/false, /*reuse_b=*/false);
}

// A fragment (16 rows x 32 k) from a row-major [rows][ld] bf16 array.
// lane<16: row=lane, k = {k0..k0+7, k0+16..k0+23}; lane>=16: row=lane-16, +8.
__device__ __forceinline__ v16bf afrag(const __bf16* arr, int row0, int ld, int k0) {
  const int lane = threadIdx.x & 31;
  const int r = lane & 15;
  const int ko = (lane >> 4) << 3;            // 0 or 8
  const __bf16* p = arr + (size_t)(row0 + r) * ld + k0 + ko;
  union { v16bf v; uint4 q[2]; } u;
  u.q[0] = *(const uint4*)p;                  // k0+ko .. +7
  u.q[1] = *(const uint4*)(p + 16);           // k0+ko+16 .. +23
  return u.v;
}

// B fragment (32 k x 16 cols) read from B^T stored row-major [cols][ld].
// lane<16: col=lane, k = {k0..k0+7, k0+16..k0+23}; lane>=16: col=lane-16, +8.
__device__ __forceinline__ v16bf bfrag(const __bf16* arrT, int col0, int ld, int k0) {
  const int lane = threadIdx.x & 31;
  const int r = lane & 15;
  const int ko = (lane >> 4) << 3;            // 0 or 8
  const __bf16* p = arrT + (size_t)(col0 + r) * ld + k0 + ko;
  union { v16bf v; uint4 q[2]; } u;
  u.q[0] = *(const uint4*)p;
  u.q[1] = *(const uint4*)(p + 16);
  return u.v;
}

// Build bf16 A-fragment directly from global f32 row-major data.
__device__ __forceinline__ v16bf afrag_f32(const float* p) {
  union { v16bf v; __bf16 e[16]; } u;
#pragma unroll
  for (int x = 0; x < 8; ++x) {
    u.e[x]     = (__bf16)p[x];
    u.e[x + 8] = (__bf16)p[x + 16];
  }
  return u.v;
}
__device__ __forceinline__ v16bf afrag_f32_mul(const float* p, const float* q) {
  union { v16bf v; __bf16 e[16]; } u;
#pragma unroll
  for (int x = 0; x < 8; ++x) {
    u.e[x]     = (__bf16)(p[x] * q[x]);
    u.e[x + 8] = (__bf16)(p[x + 16] * q[x + 16]);
  }
  return u.v;
}

// Store C-layout 16x16 f32 accumulator as bf16, row-major [rows][ld].
__device__ __forceinline__ void store_c(__bf16* dst, int rbase, int c0, int ld, v8f acc) {
  const int lane = threadIdx.x & 31;
  const int col = c0 + (lane & 15);
  const int hi  = lane >> 4;
#pragma unroll
  for (int rr = 0; rr < 8; ++rr)
    dst[(size_t)(rbase + rr + 8 * hi) * ld + col] = (__bf16)acc[rr];
}

// Store C tile transposed into VT[d][m] (ld = NN); 8 consecutive m -> b128 store.
__device__ __forceinline__ void store_c_T(__bf16* vt, int rbase, int c0, v8f acc) {
  const int lane = threadIdx.x & 31;
  const int d  = c0 + (lane & 15);
  const int hi = lane >> 4;
  union { uint4 q; __bf16 e[8]; } u;
#pragma unroll
  for (int rr = 0; rr < 8; ++rr) u.e[rr] = (__bf16)acc[rr];
  *(uint4*)(vt + (size_t)d * NN + rbase + 8 * hi) = u.q;
}

// ---------------------------------------------------------------------------
// Kernel 0: transpose + convert the 5 DxD f32 weights into bf16 W^T in d_ws.
// wt[m][o][d] = W_m[d][o];  m: 0=w_qk0 1=w_qk1 2=w_q 3=w_k 4=w_v
// ---------------------------------------------------------------------------
__global__ void prep_weights_kernel(const float* __restrict__ w_qk,
                                    const float* __restrict__ w_qkv,
                                    __bf16* __restrict__ wt) {
  const int m = blockIdx.x;
  const int o = blockIdx.y;
  const int d = threadIdx.x;
  const float* src = (m < 2) ? (w_qk + (size_t)m * DD * DD)
                             : (w_qkv + (size_t)(m - 2) * DD * DD);
  wt[((size_t)m * DD + o) * DD + d] = (__bf16)src[(size_t)d * DD + o];
}

// ---------------------------------------------------------------------------
// Kernel 1: fully fused per-batch pipeline. 1 block = 1 batch, 256 thr = 8 waves.
// ---------------------------------------------------------------------------
__global__ __launch_bounds__(256) void fused_attn_kernel(
    const float* __restrict__ feature, const float* __restrict__ indicator,
    const __bf16* __restrict__ wt, float* __restrict__ out) {
  // LDS: 144 KB total
  __shared__ __align__(16) unsigned char lds_raw[147456];
  __bf16* X0 = (__bf16*)lds_raw;                  // 64KB: T0 [256][128], then V^T [128][256]
  __bf16* X1 = (__bf16*)(lds_raw + 65536);        // 64KB: T1 [256][128], then K [256][128]
  unsigned char* GM = (unsigned char*)(lds_raw + 131072);   // 8KB gate sign mask
  __bf16* SP = (__bf16*)(lds_raw + 139264);       // 8KB: per-wave 16x32 transpose scratch

  const int b    = blockIdx.x;
  const int tid  = threadIdx.x;
  const int w    = tid >> 5;
  const int lane = tid & 31;
  const int r    = lane & 15;
  const int hi   = lane >> 4;
  const int ko   = hi << 3;
  const int rb0  = w * 32;                        // this wave owns rows rb0..rb0+31
  const float* fb = feature + (size_t)b * NN * DD;
  __bf16* myP = SP + w * 512;                     // 16x32 bf16 scratch

  // ===== Phase 1a: T0 = fuse*W0, T1 = fuse*W1 (fuse = feature*indicator) =====
  v16bf featF[2][4], fuseF[2][4];
#pragma unroll
  for (int i = 0; i < 2; ++i)
#pragma unroll
    for (int k = 0; k < 4; ++k) {
      const int row = rb0 + 16 * i + r;
      const float* p = fb + (size_t)row * DD + 32 * k + ko;
      const float* q = indicator + (size_t)row * DD + 32 * k + ko;
      featF[i][k] = afrag_f32(p);
      fuseF[i][k] = afrag_f32_mul(p, q);
    }
#pragma unroll
  for (int mat = 0; mat < 2; ++mat) {
    __bf16* dst = mat ? X1 : X0;
    const __bf16* wm = wt + (size_t)mat * DD * DD;
#pragma unroll
    for (int n = 0; n < 8; ++n) {
      v8f a0 = {}, a1 = {};
#pragma unroll
      for (int k = 0; k < 4; ++k) {
        v16bf bf = bfrag(wm, 16 * n, DD, 32 * k);
        a0 = wmma_bf16(fuseF[0][k], bf, a0);
        a1 = wmma_bf16(fuseF[1][k], bf, a1);
      }
      store_c(dst, rb0, 16 * n, DD, a0);
      store_c(dst, rb0 + 16, 16 * n, DD, a1);
    }
  }
  __syncthreads();

  // ===== Phase 2a: gate mask = sign(T0 @ T1^T)  (tanh>0 <=> arg>0) =====
#pragma unroll
  for (int i = 0; i < 2; ++i)
    for (int j = 0; j < 16; ++j) {
      v8f s = {};
#pragma unroll
      for (int k = 0; k < 4; ++k)
        s = wmma_bf16(afrag(X0, rb0 + 16 * i, DD, 32 * k),
                      bfrag(X1, 16 * j, DD, 32 * k), s);
      unsigned gbits = 0;
#pragma unroll
      for (int rr = 0; rr < 8; ++rr) gbits |= (s[rr] > 0.0f) ? (1u << rr) : 0u;
      GM[(((2 * w + i) * 16 + j) << 5) + lane] = (unsigned char)gbits;
    }
  __syncthreads();

  // ===== Phase 1b: Q (regs), K -> X1, V^T -> X0 =====
  // Q = elu(feat*Wq)+1, kept as register A-fragments via per-wave LDS bounce.
  v16bf qf[2][4];
  {
    const __bf16* wm = wt + (size_t)2 * DD * DD;
#pragma unroll
    for (int i = 0; i < 2; ++i)
#pragma unroll
      for (int k2 = 0; k2 < 4; ++k2) {          // 32-wide d-chunk = 2 n-tiles
        v8f a0 = {}, a1 = {};
#pragma unroll
        for (int k = 0; k < 4; ++k) {
          v16bf b0 = bfrag(wm, 32 * k2, DD, 32 * k);
          v16bf b1 = bfrag(wm, 32 * k2 + 16, DD, 32 * k);
          a0 = wmma_bf16(featF[i][k], b0, a0);
          a1 = wmma_bf16(featF[i][k], b1, a1);
        }
#pragma unroll
        for (int rr = 0; rr < 8; ++rr) {        // elu(x)+1
          float x0 = a0[rr]; a0[rr] = (x0 > 0.0f) ? x0 + 1.0f : __expf(x0);
          float x1 = a1[rr]; a1[rr] = (x1 > 0.0f) ? x1 + 1.0f : __expf(x1);
        }
#pragma unroll
        for (int rr = 0; rr < 8; ++rr) {        // C-layout -> row-major scratch
          myP[(rr + 8 * hi) * 32 + r]      = (__bf16)a0[rr];
          myP[(rr + 8 * hi) * 32 + 16 + r] = (__bf16)a1[rr];
        }
        asm volatile("s_wait_dscnt 0" ::: "memory");
        qf[i][k2] = afrag(myP, 0, 32, 0);
      }
  }
#pragma unroll
  for (int mat = 3; mat <= 4; ++mat) {          // K and V
    const __bf16* wm = wt + (size_t)mat * DD * DD;
#pragma unroll
    for (int n = 0; n < 8; ++n) {
      v8f a0 = {}, a1 = {};
#pragma unroll
      for (int k = 0; k < 4; ++k) {
        v16bf bf = bfrag(wm, 16 * n, DD, 32 * k);
        a0 = wmma_bf16(featF[0][k], bf, a0);
        a1 = wmma_bf16(featF[1][k], bf, a1);
      }
      if (mat == 3) {
#pragma unroll
        for (int rr = 0; rr < 8; ++rr) {
          float x0 = a0[rr]; a0[rr] = (x0 > 0.0f) ? x0 + 1.0f : __expf(x0);
          float x1 = a1[rr]; a1[rr] = (x1 > 0.0f) ? x1 + 1.0f : __expf(x1);
        }
        store_c(X1, rb0, 16 * n, DD, a0);
        store_c(X1, rb0 + 16, 16 * n, DD, a1);
      } else {
        store_c_T(X0, rb0, 16 * n, a0);         // V^T[d][m]
        store_c_T(X0, rb0 + 16, 16 * n, a1);
      }
    }
  }
  __syncthreads();

  // ===== Phase 2b: flash softmax(Q K^T) * gate, O = P @ V =====
  v8f O[2][8];
  float rmax[2][8], rsum[2][8];
#pragma unroll
  for (int i = 0; i < 2; ++i) {
#pragma unroll
    for (int n = 0; n < 8; ++n) O[i][n] = (v8f){};
#pragma unroll
    for (int rr = 0; rr < 8; ++rr) { rmax[i][rr] = -3.0e38f; rsum[i][rr] = 0.0f; }
  }

  for (int t = 0; t < 8; ++t) {                 // 32 m-columns per step
#pragma unroll
    for (int i = 0; i < 2; ++i) {
      v8f S0 = {}, S1 = {};
#pragma unroll
      for (int k = 0; k < 4; ++k) {
        S0 = wmma_bf16(qf[i][k], bfrag(X1, 32 * t,      DD, 32 * k), S0);
        S1 = wmma_bf16(qf[i][k], bfrag(X1, 32 * t + 16, DD, 32 * k), S1);
      }
      const unsigned g0 = GM[(((2 * w + i) * 16 + 2 * t) << 5) + lane];
      const unsigned g1 = GM[(((2 * w + i) * 16 + 2 * t + 1) << 5) + lane];
      v8f corrv = {};
#pragma unroll
      for (int rr = 0; rr < 8; ++rr) {
        float cmax = fmaxf(S0[rr], S1[rr]);     // row reduce over 16-lane half
#pragma unroll
        for (int m = 1; m <= 8; m <<= 1) cmax = fmaxf(cmax, __shfl_xor(cmax, m, 32));
        const float mo = rmax[i][rr];
        const float mn = fmaxf(mo, cmax);
        const float c  = __expf(mo - mn);
        const float p0 = __expf(S0[rr] - mn);
        const float p1 = __expf(S1[rr] - mn);
        float ps = p0 + p1;
#pragma unroll
        for (int m = 1; m <= 8; m <<= 1) ps += __shfl_xor(ps, m, 32);
        rmax[i][rr] = mn;
        rsum[i][rr] = rsum[i][rr] * c + ps;     // denominator is UNgated (softmax first)
        corrv[rr] = c;
        S0[rr] = ((g0 >> rr) & 1) ? p0 : 0.0f;  // numerator gets the gate
        S1[rr] = ((g1 >> rr) & 1) ? p1 : 0.0f;
      }
#pragma unroll
      for (int n = 0; n < 8; ++n) O[i][n] *= corrv;
      // P tile (16x32) -> scratch -> A fragment
#pragma unroll
      for (int rr = 0; rr < 8; ++rr) {
        myP[(rr + 8 * hi) * 32 + r]      = (__bf16)S0[rr];
        myP[(rr + 8 * hi) * 32 + 16 + r] = (__bf16)S1[rr];
      }
      asm volatile("s_wait_dscnt 0" ::: "memory");
      const v16bf pf = afrag(myP, 0, 32, 0);
#pragma unroll
      for (int n = 0; n < 8; ++n)
        O[i][n] = wmma_bf16(pf, bfrag(X0, 16 * n, NN, 32 * t), O[i][n]);
    }
  }

  // ===== Epilogue: normalize, store f32 =====
#pragma unroll
  for (int i = 0; i < 2; ++i)
#pragma unroll
    for (int n = 0; n < 8; ++n)
#pragma unroll
      for (int rr = 0; rr < 8; ++rr) {
        const int row = rb0 + 16 * i + rr + 8 * hi;
        out[((size_t)b * NN + row) * DD + 16 * n + r] = O[i][n][rr] / rsum[i][rr];
      }
}

// ---------------------------------------------------------------------------
extern "C" void kernel_launch(void* const* d_in, const int* in_sizes, int n_in,
                              void* d_out, int out_size, void* d_ws, size_t ws_size,
                              hipStream_t stream) {
  const float* feature   = (const float*)d_in[0];   // (512,256,128)
  const float* indicator = (const float*)d_in[1];   // (256,128)
  const float* w_qk      = (const float*)d_in[2];   // (2,128,128)
  const float* w_qkv     = (const float*)d_in[3];   // (3,128,128)
  float* out = (float*)d_out;
  __bf16* wt = (__bf16*)d_ws;                       // 5*128*128 bf16 = 160KB

  prep_weights_kernel<<<dim3(5, DD), DD, 0, stream>>>(w_qk, w_qkv, wt);
  fused_attn_kernel<<<512, 256, 0, stream>>>(feature, indicator, wt, out);
}
